// MultiHeadSelfAttention_84817014162130
// MI455X (gfx1250) — matrix operand order repaired and checked by reading, not compiled, not run on hardware
//
#include <hip/hip_runtime.h>
#include <hip/hip_bf16.h>

typedef __attribute__((ext_vector_type(16))) _Float16 v16h;
typedef __attribute__((ext_vector_type(8)))  _Float16 v8h;
typedef __attribute__((ext_vector_type(8)))  float    v8f;

union H16 { v16h v; v8h h[2]; };

__device__ __forceinline__ v8f wmma_f16(v16h a, v16h b, v8f c) {
    return __builtin_amdgcn_wmma_f32_16x16x32_f16(
        /*neg_a=*/false, a, /*neg_b=*/false, b,
        /*c_mod=*/(short)0, c, /*reuse_a=*/false, /*reuse_b=*/false);
}

// XOR-lane shuffle within groups of 32 via ds_swizzle (group-of-32 mode:
// offset[15]=0, xor_mask = offset[14:10], and_mask = 0x1f). Masks 1/2/4/8
// stay within each 16-lane half -> matches the WMMA C/D layout (row M lives
// in lanes 0-15 for M<8 and lanes 16-31 for M>=8).
template <int MASK>
__device__ __forceinline__ float xor_shfl(float v) {
    int i = __builtin_amdgcn_ds_swizzle(__float_as_int(v), (MASK << 10) | 0x1f);
    return __int_as_float(i);
}
__device__ __forceinline__ float rowmax16(float v) {
    v = fmaxf(v, xor_shfl<1>(v));
    v = fmaxf(v, xor_shfl<2>(v));
    v = fmaxf(v, xor_shfl<4>(v));
    v = fmaxf(v, xor_shfl<8>(v));
    return v;
}
__device__ __forceinline__ float rowsum16(float v) {
    v += xor_shfl<1>(v);
    v += xor_shfl<2>(v);
    v += xor_shfl<4>(v);
    v += xor_shfl<8>(v);
    return v;
}

// ---------------- conversion kernels ----------------

__global__ void cvt_f32_f16_kernel(const float* __restrict__ in,
                                   _Float16* __restrict__ out, int n) {
    int i = blockIdx.x * 256 + threadIdx.x;
    if (i < n) out[i] = (_Float16)in[i];
}

// in [rows, cols] fp32 -> out [cols, rows] f16
__global__ void cvt_transpose_kernel(const float* __restrict__ in,
                                     _Float16* __restrict__ out,
                                     int rows, int cols) {
    int i = blockIdx.x * 256 + threadIdx.x;
    if (i < rows * cols) {
        int r = i / cols, c = i - r * cols;
        out[c * rows + r] = (_Float16)in[i];
    }
}

// ---------------- QKV projection GEMM ----------------
// xh   : [4096, 1024] f16   (M = B*N rows)
// wt   : [3072, 1024] f16   (w_qkv transposed: [col, k])
// bias : [3072] fp32
// Outputs (f16): qh,kh = [B,H,N,D]=[2,16,2048,64]; vt = [B,H,D,N]=[2,16,64,2048]
// q is pre-scaled by 1/sqrt(D) = 0.125 (after bias) so attention skips it.
__device__ __forceinline__ void store_qkv_tile(const v8f& acc, int mrow0, int col,
                                               int lhalf,
                                               _Float16* __restrict__ qh,
                                               _Float16* __restrict__ kh,
                                               _Float16* __restrict__ vt) {
    int sel = col >> 10;          // 0=q 1=k 2=v
    int h   = (col >> 6) & 15;
    int d   = col & 63;
    float scale = (sel == 0) ? 0.125f : 1.0f;
#pragma unroll
    for (int r = 0; r < 8; ++r) {
        int tok = mrow0 + r + lhalf * 8;   // 0..4095
        int bb  = tok >> 11;
        int n   = tok & 2047;
        int bh  = bb * 16 + h;
        _Float16 val = (_Float16)(acc[r] * scale);
        if (sel == 0)      qh[(bh * 2048 + n) * 64 + d] = val;
        else if (sel == 1) kh[(bh * 2048 + n) * 64 + d] = val;
        else               vt[(bh * 64 + d) * 2048 + n] = val;
    }
}

__global__ void __launch_bounds__(256) qkv_gemm_kernel(
    const _Float16* __restrict__ xh, const _Float16* __restrict__ wt,
    const float* __restrict__ bias,
    _Float16* __restrict__ qh, _Float16* __restrict__ kh,
    _Float16* __restrict__ vt) {
    const int K = 1024;
    int warp = threadIdx.x >> 5;
    int lane = threadIdx.x & 31;
    int wid  = blockIdx.x * 8 + warp;     // 12288 waves: 128 x 96 tiles of 32x32
    int mt = wid / 96, nt = wid - mt * 96;
    int m0 = mt * 32, c0 = nt * 32;
    int l15 = lane & 15, lh = lane >> 4;

    const _Float16* arow0 = xh + (m0 + l15) * K + lh * 16;
    const _Float16* arow1 = arow0 + 16 * K;
    const _Float16* bcol0 = wt + (c0 + l15) * K + lh * 16;
    const _Float16* bcol1 = bcol0 + 16 * K;

    float bias0 = bias[c0 + l15];
    float bias1 = bias[c0 + 16 + l15];
    v8f acc00, acc01, acc10, acc11;
#pragma unroll
    for (int i = 0; i < 8; ++i) {
        acc00[i] = bias0; acc01[i] = bias1;
        acc10[i] = bias0; acc11[i] = bias1;
    }

    // Keep steady-state VGPR use low: no unrolling (occupancy hides latency).
#pragma clang loop unroll(disable)
    for (int kk = 0; kk < K; kk += 32) {
        H16 a0, a1;
        a0.h[0] = *(const v8h*)(arow0 + kk);
        a0.h[1] = *(const v8h*)(arow0 + kk + 8);
        a1.h[0] = *(const v8h*)(arow1 + kk);
        a1.h[1] = *(const v8h*)(arow1 + kk + 8);
        v16h b0 = *(const v16h*)(bcol0 + kk);
        v16h b1 = *(const v16h*)(bcol1 + kk);
        acc00 = wmma_f16(a0.v, b0, acc00);
        acc01 = wmma_f16(a0.v, b1, acc01);
        acc10 = wmma_f16(a1.v, b0, acc10);
        acc11 = wmma_f16(a1.v, b1, acc11);
    }
    store_qkv_tile(acc00, m0,      c0 + l15,      lh, qh, kh, vt);
    store_qkv_tile(acc01, m0,      c0 + 16 + l15, lh, qh, kh, vt);
    store_qkv_tile(acc10, m0 + 16, c0 + l15,      lh, qh, kh, vt);
    store_qkv_tile(acc11, m0 + 16, c0 + 16 + l15, lh, qh, kh, vt);
}

// ---------------- flash attention ----------------
// qh,kh: [32, 2048, 64] f16 (q pre-scaled), vt: [32, 64, 2048] f16
// ctx:   [B, N, C] = [2, 2048, 1024] f16 (head-interleaved, ready for proj)
__global__ void __launch_bounds__(256) attn_kernel(
    const _Float16* __restrict__ qh, const _Float16* __restrict__ kh,
    const _Float16* __restrict__ vt, _Float16* __restrict__ ctx) {
    __shared__ _Float16 lds_p[8][16 * 32];   // per-wave 16x32 P tile

    int warp = threadIdx.x >> 5;
    int lane = threadIdx.x & 31;
    int wid  = blockIdx.x * 8 + warp;        // 4096 waves: 32 bh x 128 q-tiles
    int bh = wid >> 7;
    int qt = wid & 127;
    int l15 = lane & 15, lh = lane >> 4;

    // Q A-fragments (16 rows x 32 d), kept in registers for the whole loop.
    const _Float16* qbase = qh + (bh * 2048 + qt * 16 + l15) * 64 + lh * 16;
    H16 qf0, qf1;
    qf0.h[0] = *(const v8h*)(qbase);
    qf0.h[1] = *(const v8h*)(qbase + 8);
    qf1.h[0] = *(const v8h*)(qbase + 32);
    qf1.h[1] = *(const v8h*)(qbase + 40);

    const _Float16* kbase = kh + (bh * 2048 + l15) * 64 + lh * 16;
    const _Float16* vbase = vt + (bh * 64 + l15) * 2048 + lh * 16;
    _Float16* pbuf = lds_p[warp];

    v8f o0, o1, o2, o3;
    float m_run[8], l_run[8];
#pragma unroll
    for (int r = 0; r < 8; ++r) {
        o0[r] = 0.f; o1[r] = 0.f; o2[r] = 0.f; o3[r] = 0.f;
        m_run[r] = -1e30f; l_run[r] = 0.f;
    }
    const float L2E = 1.44269504088896340736f;

#pragma clang loop unroll(disable)
    for (int kb = 0; kb < 2048; kb += 32) {
        // S = Q @ K^T for 32 keys (two 16-key tiles), d reduced in 2 steps.
        v8f s0, s1;
#pragma unroll
        for (int r = 0; r < 8; ++r) { s0[r] = 0.f; s1[r] = 0.f; }
        s0 = wmma_f16(qf0.v, *(const v16h*)(kbase + kb * 64),             s0);
        s1 = wmma_f16(qf0.v, *(const v16h*)(kbase + (kb + 16) * 64),      s1);
        s0 = wmma_f16(qf1.v, *(const v16h*)(kbase + kb * 64 + 32),        s0);
        s1 = wmma_f16(qf1.v, *(const v16h*)(kbase + (kb + 16) * 64 + 32), s1);

        float alpha[8];
#pragma unroll
        for (int r = 0; r < 8; ++r) {
            float rm = rowmax16(fmaxf(s0[r], s1[r]));
            float mn = fmaxf(m_run[r], rm);
            alpha[r] = exp2f((m_run[r] - mn) * L2E);
            float p0 = exp2f((s0[r] - mn) * L2E);
            float p1 = exp2f((s1[r] - mn) * L2E);
            l_run[r] = l_run[r] * alpha[r] + rowsum16(p0 + p1);
            m_run[r] = mn;
            int row = r + lh * 8;
            pbuf[row * 32 + l15]      = (_Float16)p0;
            pbuf[row * 32 + 16 + l15] = (_Float16)p1;
        }
#pragma unroll
        for (int r = 0; r < 8; ++r) {
            o0[r] *= alpha[r]; o1[r] *= alpha[r];
            o2[r] *= alpha[r]; o3[r] *= alpha[r];
        }
        // Same-wave LDS ordering: wait for our 16x32 P stores, then reload in
        // A-fragment layout (plain strided reads == A layout, no transpose).
        asm volatile("s_wait_dscnt 0" ::: "memory");
        H16 pf;
        pf.h[0] = *(const v8h*)(pbuf + l15 * 32 + lh * 16);
        pf.h[1] = *(const v8h*)(pbuf + l15 * 32 + lh * 16 + 8);
        asm volatile("" ::: "memory");  // keep next iter's stores after these loads

        o0 = wmma_f16(pf.v, *(const v16h*)(vbase + kb),             o0);
        o1 = wmma_f16(pf.v, *(const v16h*)(vbase + 16 * 2048 + kb), o1);
        o2 = wmma_f16(pf.v, *(const v16h*)(vbase + 32 * 2048 + kb), o2);
        o3 = wmma_f16(pf.v, *(const v16h*)(vbase + 48 * 2048 + kb), o3);
    }

    int b = bh >> 4, h = bh & 15;
    _Float16* cbase = ctx + (size_t)(b * 2048 + qt * 16) * 1024 + h * 64;
#pragma unroll
    for (int r = 0; r < 8; ++r) {
        float inv = 1.0f / l_run[r];
        _Float16* rowp = cbase + (r + lh * 8) * 1024;
        rowp[l15]      = (_Float16)(o0[r] * inv);
        rowp[16 + l15] = (_Float16)(o1[r] * inv);
        rowp[32 + l15] = (_Float16)(o2[r] * inv);
        rowp[48 + l15] = (_Float16)(o3[r] * inv);
    }
}

// ---------------- output projection GEMM ----------------
// ctx [4096,1024] f16, wt = w_proj^T [1024,1024] f16, out [4096,1024] fp32
__global__ void __launch_bounds__(256) proj_gemm_kernel(
    const _Float16* __restrict__ ctx, const _Float16* __restrict__ wt,
    const float* __restrict__ bias, float* __restrict__ out) {
    const int K = 1024;
    int warp = threadIdx.x >> 5;
    int lane = threadIdx.x & 31;
    int wid  = blockIdx.x * 8 + warp;   // 4096 waves: 128 x 32 tiles of 32x32
    int mt = wid >> 5, nt = wid & 31;
    int m0 = mt * 32, c0 = nt * 32;
    int l15 = lane & 15, lh = lane >> 4;

    const _Float16* arow0 = ctx + (m0 + l15) * K + lh * 16;
    const _Float16* arow1 = arow0 + 16 * K;
    const _Float16* bcol0 = wt + (c0 + l15) * K + lh * 16;
    const _Float16* bcol1 = bcol0 + 16 * K;

    float bias0 = bias[c0 + l15];
    float bias1 = bias[c0 + 16 + l15];
    v8f acc00, acc01, acc10, acc11;
#pragma unroll
    for (int i = 0; i < 8; ++i) {
        acc00[i] = bias0; acc01[i] = bias1;
        acc10[i] = bias0; acc11[i] = bias1;
    }

#pragma clang loop unroll(disable)
    for (int kk = 0; kk < K; kk += 32) {
        H16 a0, a1;
        a0.h[0] = *(const v8h*)(arow0 + kk);
        a0.h[1] = *(const v8h*)(arow0 + kk + 8);
        a1.h[0] = *(const v8h*)(arow1 + kk);
        a1.h[1] = *(const v8h*)(arow1 + kk + 8);
        v16h b0 = *(const v16h*)(bcol0 + kk);
        v16h b1 = *(const v16h*)(bcol1 + kk);
        acc00 = wmma_f16(a0.v, b0, acc00);
        acc01 = wmma_f16(a0.v, b1, acc01);
        acc10 = wmma_f16(a1.v, b0, acc10);
        acc11 = wmma_f16(a1.v, b1, acc11);
    }
#pragma unroll
    for (int r = 0; r < 8; ++r) {
        int tok0 = m0 + r + lh * 8;
        out[tok0 * 1024 + c0 + l15]             = acc00[r];
        out[tok0 * 1024 + c0 + 16 + l15]        = acc01[r];
        out[(tok0 + 16) * 1024 + c0 + l15]      = acc10[r];
        out[(tok0 + 16) * 1024 + c0 + 16 + l15] = acc11[r];
    }
}

extern "C" void kernel_launch(void* const* d_in, const int* in_sizes, int n_in,
                              void* d_out, int out_size, void* d_ws, size_t ws_size,
                              hipStream_t stream) {
    (void)in_sizes; (void)n_in; (void)out_size; (void)ws_size;
    const float* x      = (const float*)d_in[0];
    const float* w_qkv  = (const float*)d_in[1];
    const float* b_qkv  = (const float*)d_in[2];
    const float* w_proj = (const float*)d_in[3];
    const float* b_proj = (const float*)d_in[4];
    float* out = (float*)d_out;

    char* ws = (char*)d_ws;
    _Float16* x_h     = (_Float16*)(ws);                 //  8 MB  [4096,1024]
    _Float16* wqkv_t  = (_Float16*)(ws + (8u  << 20));   //  6 MB  [3072,1024]
    _Float16* wproj_t = (_Float16*)(ws + (14u << 20));   //  2 MB  [1024,1024]
    _Float16* q_h     = (_Float16*)(ws + (16u << 20));   //  8 MB  [32,2048,64]
    _Float16* k_h     = (_Float16*)(ws + (24u << 20));   //  8 MB  [32,2048,64]
    _Float16* v_t     = (_Float16*)(ws + (32u << 20));   //  8 MB  [32,64,2048]
    _Float16* ctx_h   = (_Float16*)(ws + (40u << 20));   //  8 MB  [4096,1024]

    cvt_f32_f16_kernel<<<(4194304 + 255) / 256, 256, 0, stream>>>(x, x_h, 4194304);
    cvt_transpose_kernel<<<(3145728 + 255) / 256, 256, 0, stream>>>(w_qkv, wqkv_t, 1024, 3072);
    cvt_transpose_kernel<<<(1048576 + 255) / 256, 256, 0, stream>>>(w_proj, wproj_t, 1024, 1024);

    qkv_gemm_kernel<<<1536, 256, 0, stream>>>(x_h, wqkv_t, b_qkv, q_h, k_h, v_t);
    attn_kernel<<<512, 256, 0, stream>>>(q_h, k_h, v_t, ctx_h);
    proj_gemm_kernel<<<512, 256, 0, stream>>>(ctx_h, wproj_t, b_proj, out);
}
